// RelPartialLearnableDecoderLayer_6253472383719
// MI455X (gfx1250) — compile-verified, matrix operand order checked by reading, not run on hardware
//
#include <hip/hip_runtime.h>
#include <hip/hip_bf16.h>

// ---------------------------------------------------------------------------
// Transformer-XL RelPartialLearnableDecoderLayer for MI455X (gfx1250).
// All contractions run through v_wmma_f32_16x16x32_bf16 (wave32).
// Wave tile = 64x32 (4x2 fragments -> 64 accumulator VGPRs, no spills).
// All GEMMs use the NT contiguous-B path (weights pre-transposed once,
// V packed transposed once) -> fragment loads are 2x global_load_b128.
// ---------------------------------------------------------------------------

typedef __bf16 bf16_t;
typedef __attribute__((ext_vector_type(16))) __bf16 v16bf;
typedef __attribute__((ext_vector_type(8)))  float  v8f;

#define TXL_QLEN   1024
#define TXL_MLEN   1024
#define TXL_KLEN   2048
#define TXL_BSZ    4
#define TXL_DMODEL 1024
#define TXL_NHEAD  16
#define TXL_DHEAD  64
#define TXL_DINNER 4096
#define TXL_NHD    1024
#define TXL_ROWS   (TXL_QLEN * TXL_BSZ)      /* 4096 */
#define TXL_CROWS  (TXL_KLEN * TXL_BSZ)      /* 8192 */
#define TXL_SLICES (TXL_BSZ * TXL_NHEAD)     /* 64 */

// --------------------------- fragment loader -------------------------------
// 16-bit A-matrix 16x32 layout (ISA 7.12.2): lane = half*16 + m;
//   vector elem e<8  -> K = k0 + 8*half + e          (contiguous 16 bytes)
//   vector elem e>=8 -> K = k0 + 16 + 8*half + (e-8) (contiguous 16 bytes)
// B operand mirrors A with lane = half*16 + n (so we feed it rows of B^T).

__device__ __forceinline__ v16bf txl_frag_contig(const bf16_t* __restrict__ p, int half) {
  union { uint4 u[2]; v16bf v; } t;
  t.u[0] = *reinterpret_cast<const uint4*>(p + 8 * half);
  t.u[1] = *reinterpret_cast<const uint4*>(p + 16 + 8 * half);
  return t.v;
}

// --------------------------- WMMA GEMM (NT) --------------------------------
// C[M,N] = A[M,K] * B^T,  A row-major [M,K] (lda), Bt row-major [N,K] (ldb).
// One wave -> 64x32 output tile.  Output f32 (Cf) or bf16 (Cb), one null.
__global__ __launch_bounds__(256)
void txl_wmma_gemm_nt(const bf16_t* __restrict__ A, const bf16_t* __restrict__ Bt,
                      float* __restrict__ Cf, bf16_t* __restrict__ Cb,
                      int M, int N, int K, int lda, int ldb, int ldc) {
  const int wave   = blockIdx.x * (blockDim.x >> 5) + (threadIdx.x >> 5);
  const int lane   = threadIdx.x & 31;
  const int tilesN = N >> 5;
  const int tm     = (wave / tilesN) << 6;
  const int tn     = (wave % tilesN) << 5;
  if (tm >= M) return;

  const int half = lane >> 4;   // K-half selector
  const int mr   = lane & 15;   // row (A) / col (B,C) within 16

  v8f acc[4][2];
#pragma unroll
  for (int mi = 0; mi < 4; ++mi)
#pragma unroll
    for (int ni = 0; ni < 2; ++ni)
#pragma unroll
      for (int e = 0; e < 8; ++e) acc[mi][ni][e] = 0.0f;

  const bf16_t* arow = A  + (size_t)(tm + mr) * lda;
  const bf16_t* brow = Bt + (size_t)(tn + mr) * ldb;

  for (int k0 = 0; k0 < K; k0 += 32) {
    if (k0 + 32 < K)  // stream next A chunk (global_prefetch_b8)
      __builtin_prefetch(arow + k0 + 32, 0, 1);

    v16bf b0 = txl_frag_contig(brow + k0, half);
    v16bf b1 = txl_frag_contig(brow + (size_t)16 * ldb + k0, half);
#pragma unroll
    for (int mi = 0; mi < 4; ++mi) {
      v16bf a = txl_frag_contig(arow + (size_t)mi * 16 * lda + k0, half);
      acc[mi][0] = __builtin_amdgcn_wmma_f32_16x16x32_bf16(
          false, a, false, b0, (short)0, acc[mi][0], false, false);
      acc[mi][1] = __builtin_amdgcn_wmma_f32_16x16x32_bf16(
          false, a, false, b1, (short)0, acc[mi][1], false, false);
    }
  }

  // C layout: lane (half,mr): vector elem e -> m_local = 8*half+e, n_local = mr
#pragma unroll
  for (int mi = 0; mi < 4; ++mi)
#pragma unroll
    for (int ni = 0; ni < 2; ++ni)
#pragma unroll
      for (int e = 0; e < 8; ++e) {
        const size_t m = (size_t)(tm + mi * 16 + 8 * half + e);
        const size_t n = (size_t)(tn + ni * 16 + mr);
        const float  v = acc[mi][ni][e];
        if (Cb) Cb[m * ldc + n] = (bf16_t)v;
        else    Cf[m * ldc + n] = v;
      }
}

// --------------------------- small helper kernels --------------------------

__global__ void txl_cast_kernel(const float* __restrict__ src, bf16_t* __restrict__ dst, size_t n) {
  size_t i = (size_t)blockIdx.x * blockDim.x + threadIdx.x;
  if (i < n) dst[i] = (bf16_t)src[i];
}

// dst[c][r] = bf16(src[r][c])  (rows x cols -> cols x rows)
__global__ void txl_cast_transpose_kernel(const float* __restrict__ src, bf16_t* __restrict__ dst,
                                          int rows, int cols) {
  size_t i = (size_t)blockIdx.x * blockDim.x + threadIdx.x;
  if (i >= (size_t)rows * cols) return;
  const int r = (int)(i / cols);
  const int c = (int)(i % cols);
  dst[(size_t)c * rows + r] = (bf16_t)src[i];
}

// q/k/v live in w_heads (bf16, [8192 x 3072], row = j*BSZ+b).  Build biased Q
// slices qw,qr laid out [slice s=b*16+h][i][d] (lda = 64).
__global__ void txl_pack_q_kernel(const bf16_t* __restrict__ WH,
                                  const float* __restrict__ rwb, const float* __restrict__ rrb,
                                  bf16_t* __restrict__ qw, bf16_t* __restrict__ qr) {
  size_t idx = (size_t)blockIdx.x * blockDim.x + threadIdx.x;
  if (idx >= (size_t)TXL_SLICES * TXL_QLEN * TXL_DHEAD) return;
  const int d  = idx & 63;
  const int i  = (int)((idx >> 6) & 1023);
  const int s  = (int)(idx >> 16);
  const int bi = s >> 4;
  const int hn = s & 15;
  const float q = (float)WH[(size_t)((TXL_MLEN + i) * TXL_BSZ + bi) * 3072 + hn * 64 + d];
  qw[idx] = (bf16_t)(q + rwb[hn * 64 + d]);
  qr[idx] = (bf16_t)(q + rrb[hn * 64 + d]);
}

// vT[s][d][j] = V[j][b][h][d] so attn_vec GEMM can use the NT path.
__global__ void txl_pack_vT_kernel(const bf16_t* __restrict__ WH, bf16_t* __restrict__ vT) {
  size_t idx = (size_t)blockIdx.x * blockDim.x + threadIdx.x;
  if (idx >= (size_t)TXL_SLICES * TXL_DHEAD * TXL_KLEN) return;
  const int j  = (int)(idx & (TXL_KLEN - 1));
  const int d  = (int)((idx >> 11) & 63);
  const int s  = (int)(idx >> 17);
  const int bi = s >> 4;
  const int hn = s & 15;
  vT[idx] = WH[(size_t)(j * TXL_BSZ + bi) * 3072 + 2 * TXL_NHD + hn * 64 + d];
}

// scores: AC[i][j] + rel_shift(BD)[i][j] = AC[i][j] + BD[i][j+QLEN-1-i],
// causal mask j <= i+MLEN, softmax over j, write bf16 probabilities.
__global__ void txl_softmax_kernel(const float* __restrict__ S, const float* __restrict__ BD,
                                   bf16_t* __restrict__ P) {
  const int    row  = blockIdx.x;          // row = slice*1024 + i
  const int    i    = row & (TXL_QLEN - 1);
  const float* srow = S  + (size_t)row * TXL_KLEN;
  const float* brow = BD + (size_t)row * TXL_KLEN;
  bf16_t*      prow = P  + (size_t)row * TXL_KLEN;
  const int    jmax = i + TXL_MLEN;        // inclusive; always < KLEN
  const int    sh   = TXL_QLEN - 1 - i;
  const float  scale = 0.125f;             // 1/sqrt(64)

  __shared__ float red[256];
  const int t = threadIdx.x;

  float lmax = -3.4e38f;
  for (int j = t; j <= jmax; j += 256)
    lmax = fmaxf(lmax, (srow[j] + brow[j + sh]) * scale);
  red[t] = lmax; __syncthreads();
  for (int s2 = 128; s2 > 0; s2 >>= 1) { if (t < s2) red[t] = fmaxf(red[t], red[t + s2]); __syncthreads(); }
  const float m = red[0]; __syncthreads();

  float lsum = 0.0f;
  for (int j = t; j <= jmax; j += 256)
    lsum += __expf((srow[j] + brow[j + sh]) * scale - m);
  red[t] = lsum; __syncthreads();
  for (int s2 = 128; s2 > 0; s2 >>= 1) { if (t < s2) red[t] += red[t + s2]; __syncthreads(); }
  const float inv = 1.0f / red[0];

  for (int j = t; j < TXL_KLEN; j += 256) {
    float p = 0.0f;
    if (j <= jmax) p = __expf((srow[j] + brow[j + sh]) * scale - m) * inv;
    prow[j] = (bf16_t)p;
  }
}

__global__ void txl_bias_relu_kernel(const float* __restrict__ H, const float* __restrict__ b1,
                                     bf16_t* __restrict__ Hb, int ncols, size_t n) {
  size_t i = (size_t)blockIdx.x * blockDim.x + threadIdx.x;
  if (i >= n) return;
  float v = H[i] + b1[i % ncols];
  Hb[i] = (bf16_t)fmaxf(v, 0.0f);
}

// Of = LayerNorm(X + Y [+ eb]) * gw + gb ; optionally also bf16 copy (Ob).
__global__ void txl_add_ln_kernel(const float* __restrict__ X, const float* __restrict__ Y,
                                  const float* __restrict__ eb,
                                  const float* __restrict__ gw, const float* __restrict__ gb,
                                  float* __restrict__ Of, bf16_t* __restrict__ Ob) {
  const int row = blockIdx.x;                 // 4096 rows of 1024
  const float* x = X + (size_t)row * TXL_DMODEL;
  const float* y = Y + (size_t)row * TXL_DMODEL;
  __shared__ float red[256];
  const int t = threadIdx.x;

  float v[4], s = 0.0f;
#pragma unroll
  for (int q = 0; q < 4; ++q) {
    const int c = t + q * 256;
    float a = x[c] + y[c];
    if (eb) a += eb[c];
    v[q] = a; s += a;
  }
  red[t] = s; __syncthreads();
  for (int s2 = 128; s2 > 0; s2 >>= 1) { if (t < s2) red[t] += red[t + s2]; __syncthreads(); }
  const float mu = red[0] * (1.0f / TXL_DMODEL); __syncthreads();

  float s2v = 0.0f;
#pragma unroll
  for (int q = 0; q < 4; ++q) { const float d = v[q] - mu; s2v += d * d; }
  red[t] = s2v; __syncthreads();
  for (int s2 = 128; s2 > 0; s2 >>= 1) { if (t < s2) red[t] += red[t + s2]; __syncthreads(); }
  const float rstd = rsqrtf(red[0] * (1.0f / TXL_DMODEL) + 1e-5f);

#pragma unroll
  for (int q = 0; q < 4; ++q) {
    const int c = t + q * 256;
    const float o = (v[q] - mu) * rstd * gw[c] + gb[c];
    Of[(size_t)row * TXL_DMODEL + c] = o;
    if (Ob) Ob[(size_t)row * TXL_DMODEL + c] = (bf16_t)o;
  }
}

// ------------------------------- host side ---------------------------------

extern "C" void kernel_launch(void* const* d_in, const int* in_sizes, int n_in,
                              void* d_out, int out_size, void* d_ws, size_t ws_size,
                              hipStream_t stream) {
  (void)in_sizes; (void)n_in; (void)out_size; (void)ws_size;

  const float* dec_inp  = (const float*)d_in[0];
  const float* r_pos    = (const float*)d_in[1];
  const float* r_w_bias = (const float*)d_in[2];
  const float* r_r_bias = (const float*)d_in[3];
  const float* mems     = (const float*)d_in[4];
  /* d_in[5] = dec_attn_mask : causal mask reproduced analytically */
  const float* qkv_w    = (const float*)d_in[6];
  const float* r_w      = (const float*)d_in[7];
  const float* o_w      = (const float*)d_in[8];
  const float* ln1_w    = (const float*)d_in[9];
  const float* ln1_b    = (const float*)d_in[10];
  const float* ln2_w    = (const float*)d_in[11];
  const float* ln2_b    = (const float*)d_in[12];
  const float* ff_w1    = (const float*)d_in[13];
  const float* ff_b1    = (const float*)d_in[14];
  const float* ff_w2    = (const float*)d_in[15];
  const float* ff_b2    = (const float*)d_in[16];
  float* out = (float*)d_out;

  char* ws = (char*)d_ws;
  auto alloc = [&](size_t bytes) -> char* {
    char* p = ws;
    ws += (bytes + 255) & ~(size_t)255;
    return p;
  };

  bf16_t* cat_bf   = (bf16_t*)alloc((size_t)TXL_CROWS * TXL_DMODEL * 2);
  bf16_t* qkvwT    = (bf16_t*)alloc((size_t)TXL_DMODEL * 3 * TXL_NHD * 2);   // [3072][1024]
  bf16_t* rwT      = (bf16_t*)alloc((size_t)TXL_DMODEL * TXL_NHD * 2);       // [1024][1024]
  bf16_t* owT      = (bf16_t*)alloc((size_t)TXL_NHD * TXL_DMODEL * 2);       // [1024][1024]
  bf16_t* ffw1T    = (bf16_t*)alloc((size_t)TXL_DMODEL * TXL_DINNER * 2);    // [4096][1024]
  bf16_t* ffw2T    = (bf16_t*)alloc((size_t)TXL_DINNER * TXL_DMODEL * 2);    // [1024][4096]
  bf16_t* r_bf     = (bf16_t*)alloc((size_t)TXL_KLEN * TXL_DMODEL * 2);
  bf16_t* wheads   = (bf16_t*)alloc((size_t)TXL_CROWS * 3 * TXL_NHD * 2);
  bf16_t* rk_bf    = (bf16_t*)alloc((size_t)TXL_KLEN * TXL_NHD * 2);
  bf16_t* qw_bf    = (bf16_t*)alloc((size_t)TXL_SLICES * TXL_QLEN * TXL_DHEAD * 2);
  bf16_t* qr_bf    = (bf16_t*)alloc((size_t)TXL_SLICES * TXL_QLEN * TXL_DHEAD * 2);
  bf16_t* vT_bf    = (bf16_t*)alloc((size_t)TXL_SLICES * TXL_DHEAD * TXL_KLEN * 2);
  float*  Sbuf     = (float*) alloc((size_t)TXL_SLICES * TXL_QLEN * TXL_KLEN * 4);
  float*  Bbuf     = (float*) alloc((size_t)TXL_SLICES * TXL_QLEN * TXL_KLEN * 4);
  bf16_t* Pbuf     = (bf16_t*)alloc((size_t)TXL_SLICES * TXL_QLEN * TXL_KLEN * 2);
  bf16_t* av_bf    = (bf16_t*)alloc((size_t)TXL_ROWS * TXL_NHD * 2);
  float*  attnout  = (float*) alloc((size_t)TXL_ROWS * TXL_DMODEL * 4);
  float*  out1_f   = (float*) alloc((size_t)TXL_ROWS * TXL_DMODEL * 4);
  bf16_t* out1_bf  = (bf16_t*)alloc((size_t)TXL_ROWS * TXL_DMODEL * 2);
  float*  h_f      = (float*) alloc((size_t)TXL_ROWS * TXL_DINNER * 4);
  bf16_t* h_bf     = (bf16_t*)alloc((size_t)TXL_ROWS * TXL_DINNER * 2);
  float*  ff_f     = (float*) alloc((size_t)TXL_ROWS * TXL_DMODEL * 4);

  auto cast = [&](const float* src, bf16_t* dst, size_t n) {
    txl_cast_kernel<<<(unsigned)((n + 255) / 256), 256, 0, stream>>>(src, dst, n);
  };
  auto castT = [&](const float* src, bf16_t* dst, int rows, int cols) {
    const size_t n = (size_t)rows * cols;
    txl_cast_transpose_kernel<<<(unsigned)((n + 255) / 256), 256, 0, stream>>>(src, dst, rows, cols);
  };
  auto gemm = [&](const bf16_t* A, const bf16_t* Bt, float* Cf, bf16_t* Cb,
                  int M, int N, int K, int lda, int ldb, int ldc) {
    const int waves  = (M / 64) * (N / 32);
    const int blocks = (waves + 7) / 8;        // 8 waves (256 thr) per block
    txl_wmma_gemm_nt<<<blocks, 256, 0, stream>>>(A, Bt, Cf, Cb, M, N, K, lda, ldb, ldc);
  };

  // ---- stage 0: down-convert (and transpose weights) -----------------------
  const size_t half_cat = (size_t)TXL_MLEN * TXL_BSZ * TXL_DMODEL;   // 4M
  cast(mems,    cat_bf,            half_cat);        // cat = [mems; dec_inp]
  cast(dec_inp, cat_bf + half_cat, half_cat);
  cast(r_pos, r_bf, (size_t)TXL_KLEN * TXL_DMODEL);
  castT(qkv_w, qkvwT, TXL_DMODEL, 3 * TXL_NHD);
  castT(r_w,   rwT,   TXL_DMODEL, TXL_NHD);
  castT(o_w,   owT,   TXL_NHD,    TXL_DMODEL);
  castT(ff_w1, ffw1T, TXL_DMODEL, TXL_DINNER);
  castT(ff_w2, ffw2T, TXL_DINNER, TXL_DMODEL);

  // ---- stage 1: projections ------------------------------------------------
  // w_heads = cat @ qkv_w   (8192 x 3072, K=1024)
  gemm(cat_bf, qkvwT, nullptr, wheads, TXL_CROWS, 3 * TXL_NHD, TXL_DMODEL,
       TXL_DMODEL, TXL_DMODEL, 3 * TXL_NHD);
  // r_head_k = r @ r_w      (2048 x 1024, K=1024)
  gemm(r_bf, rwT, nullptr, rk_bf, TXL_KLEN, TXL_NHD, TXL_DMODEL,
       TXL_DMODEL, TXL_DMODEL, TXL_NHD);
  // biased Q slices + transposed V slices
  {
    const size_t nq = (size_t)TXL_SLICES * TXL_QLEN * TXL_DHEAD;
    txl_pack_q_kernel<<<(unsigned)((nq + 255) / 256), 256, 0, stream>>>(
        wheads, r_w_bias, r_r_bias, qw_bf, qr_bf);
    const size_t nv = (size_t)TXL_SLICES * TXL_DHEAD * TXL_KLEN;
    txl_pack_vT_kernel<<<(unsigned)((nv + 255) / 256), 256, 0, stream>>>(wheads, vT_bf);
  }

  // ---- stage 2: attention scores (per (b,head) slice) ----------------------
  for (int s = 0; s < TXL_SLICES; ++s) {
    const int bi = s >> 4, hn = s & 15;
    // AC = Q_bn (1024x64) @ K_bn^T ; K rows read in-place from w_heads (NT)
    gemm(qw_bf + (size_t)s * TXL_QLEN * TXL_DHEAD,
         wheads + (size_t)bi * 3072 + TXL_NHD + hn * 64,
         Sbuf + (size_t)s * TXL_QLEN * TXL_KLEN, nullptr,
         TXL_QLEN, TXL_KLEN, TXL_DHEAD, TXL_DHEAD, TXL_BSZ * 3072, TXL_KLEN);
    // BD = Qr_bn (1024x64) @ r_head_k_n^T (NT)
    gemm(qr_bf + (size_t)s * TXL_QLEN * TXL_DHEAD,
         rk_bf + hn * 64,
         Bbuf + (size_t)s * TXL_QLEN * TXL_KLEN, nullptr,
         TXL_QLEN, TXL_KLEN, TXL_DHEAD, TXL_DHEAD, TXL_NHD, TXL_KLEN);
  }

  // rel-shift + mask + softmax -> bf16 probabilities
  txl_softmax_kernel<<<TXL_SLICES * TXL_QLEN, 256, 0, stream>>>(Sbuf, Bbuf, Pbuf);

  // attn_vec = P (1024x2048) @ V_bn (2048x64) ; vT gives NT path
  for (int s = 0; s < TXL_SLICES; ++s) {
    const int bi = s >> 4, hn = s & 15;
    gemm(Pbuf + (size_t)s * TXL_QLEN * TXL_KLEN,
         vT_bf + (size_t)s * TXL_DHEAD * TXL_KLEN,
         nullptr, av_bf + (size_t)bi * TXL_NHD + hn * 64,
         TXL_QLEN, TXL_DHEAD, TXL_KLEN, TXL_KLEN, TXL_KLEN, TXL_BSZ * TXL_NHD);
  }

  // ---- stage 3: output projection + LN1 ------------------------------------
  gemm(av_bf, owT, attnout, nullptr, TXL_ROWS, TXL_DMODEL, TXL_NHD,
       TXL_NHD, TXL_NHD, TXL_DMODEL);
  txl_add_ln_kernel<<<TXL_ROWS, 256, 0, stream>>>(attnout, dec_inp, nullptr,
                                                  ln1_w, ln1_b, out1_f, out1_bf);

  // ---- stage 4: FFN + LN2 --------------------------------------------------
  gemm(out1_bf, ffw1T, h_f, nullptr, TXL_ROWS, TXL_DINNER, TXL_DMODEL,
       TXL_DMODEL, TXL_DMODEL, TXL_DINNER);
  {
    const size_t n = (size_t)TXL_ROWS * TXL_DINNER;
    txl_bias_relu_kernel<<<(unsigned)((n + 255) / 256), 256, 0, stream>>>(
        h_f, ff_b1, h_bf, TXL_DINNER, n);
  }
  gemm(h_bf, ffw2T, ff_f, nullptr, TXL_ROWS, TXL_DMODEL, TXL_DINNER,
       TXL_DINNER, TXL_DINNER, TXL_DMODEL);
  txl_add_ln_kernel<<<TXL_ROWS, 256, 0, stream>>>(ff_f, out1_f, ff_b2,
                                                  ln2_w, ln2_b, out, nullptr);
}
